// GraphAttentionLayer_21603685499400
// MI455X (gfx1250) — matrix operand order repaired and checked by reading, not compiled, not run on hardware
//
#include <hip/hip_runtime.h>
#include <hip/hip_bf16.h>

// ---- CDNA5 WMMA types ----
typedef __attribute__((ext_vector_type(16))) __bf16 v16bf;
typedef __attribute__((ext_vector_type(8)))  __bf16 v8bf;
typedef __attribute__((ext_vector_type(16))) float  v16f;
typedef __attribute__((ext_vector_type(8)))  float  v8f;

union BF16x16 {       // one bf16 WMMA A/B operand (8 VGPRs)
    v16bf v;
    uint4 q[2];
};
union BF16x8 {        // 8 packed bf16 (4 VGPRs) for stores
    v8bf v;
    uint4 q;
};

#define BATCH 8
#define NNODE 2048
#define FIN   128
#define FOUT  64
#define ALPHA 0.2f
#define NEG_BIG (-9.0e15f)
#define LOG2E 1.4426950408889634f

#if __has_builtin(__builtin_amdgcn_exp2f)
#define EXP2F __builtin_amdgcn_exp2f
#else
#define EXP2F exp2f
#endif

__device__ __forceinline__ v8f zero8() {
    v8f z = {0.f, 0.f, 0.f, 0.f, 0.f, 0.f, 0.f, 0.f};
    return z;
}

// =====================================================================
// Kernel 0: one-shot W (64x128 f32) -> bf16, row-major. 16 KB -> L2.
// =====================================================================
__global__ void __launch_bounds__(256)
gat_cvtw_kernel(const float* __restrict__ W, __bf16* __restrict__ Wbf)
{
    const int i = (blockIdx.x * blockDim.x + threadIdx.x) * 2;
    if (i < FOUT * FIN) {
        Wbf[i]     = (__bf16)W[i];
        Wbf[i + 1] = (__bf16)W[i + 1];
    }
}

// =====================================================================
// Kernel 1: Wh = x @ W^T  (bf16 WMMA, f32 accum).
// Stores Wh transposed (feature-major) in bf16 for kernel 2's B operand,
// and computes f1 = Wh@a1, f2 = Wh@a2 via cross-lane reductions.
// One wave handles a 16-row tile. 1024 waves total.
// =====================================================================
__global__ void __launch_bounds__(256)
gat_wh_kernel(const float* __restrict__ x, const __bf16* __restrict__ Wbf,
              const float* __restrict__ a,
              __bf16* __restrict__ WhT,   // [B*64][2048] bf16
              float* __restrict__ f1, float* __restrict__ f2)
{
    const int lane = threadIdx.x & 31;
    const int wave = threadIdx.x >> 5;
    const int tile = blockIdx.x * (blockDim.x >> 5) + wave;   // 0..1023
    const int r0   = tile * 16;          // flattened row base (b*N + n)
    const int b    = r0 >> 11;
    const int n0   = r0 & (NNODE - 1);
    const int mrow = lane & 15;
    const int hi   = lane >> 4;

    v8f acc[4];
#pragma unroll
    for (int ot = 0; ot < 4; ++ot) acc[ot] = zero8();

#pragma unroll
    for (int s = 0; s < 4; ++s) {        // K steps of 32 over Fin=128
        const int k0 = s * 32;
        // A fragment: 16x32 bf16. lanes 0-15: K k0+0..7 / k0+16..23 ;
        // lanes 16-31: K k0+8..15 / k0+24..31 (pair-packed per VGPR).
        const float* xr = x + (size_t)(r0 + mrow) * FIN + k0 + hi * 8;
        const float4 x0 = *(const float4*)(xr);
        const float4 x1 = *(const float4*)(xr + 4);
        const float4 x2 = *(const float4*)(xr + 16);
        const float4 x3 = *(const float4*)(xr + 20);
        v16f af;
        af[0]  = x0.x; af[1]  = x0.y; af[2]  = x0.z; af[3]  = x0.w;
        af[4]  = x1.x; af[5]  = x1.y; af[6]  = x1.z; af[7]  = x1.w;
        af[8]  = x2.x; af[9]  = x2.y; af[10] = x2.z; af[11] = x2.w;
        af[12] = x3.x; af[13] = x3.y; af[14] = x3.z; af[15] = x3.w;
        const v16bf A = __builtin_convertvector(af, v16bf);

#pragma unroll
        for (int ot = 0; ot < 4; ++ot) {
            // B fragment: 32x16 bf16, K contiguous in pre-converted Wbf.
            const int o = mrow + ot * 16;
            const __bf16* wp = Wbf + (size_t)o * FIN + k0 + hi * 8;
            BF16x16 Bf;
            Bf.q[0] = *(const uint4*)wp;
            Bf.q[1] = *(const uint4*)(wp + 16);
            acc[ot] = __builtin_amdgcn_wmma_f32_16x16x32_bf16(
                false, A, false, Bf.v, (short)0, acc[ot], false, false);
        }
    }

    // ---- store Wh transposed as bf16: WhT[(b*64+o)][n], n contiguous ----
#pragma unroll
    for (int ot = 0; ot < 4; ++ot) {
        const int o = mrow + ot * 16;
        BF16x8 st;
        st.v = __builtin_convertvector(acc[ot], v8bf);
        *(uint4*)(WhT + ((size_t)(b * FOUT + o)) * NNODE + n0 + hi * 8) = st.q;
    }

    // ---- f1/f2: per-row dot with a1,a2; reduce over the 16 lanes/half ----
    float a1v[4], a2v[4];
#pragma unroll
    for (int ot = 0; ot < 4; ++ot) {
        a1v[ot] = a[mrow + 16 * ot];
        a2v[ot] = a[FOUT + mrow + 16 * ot];
    }
    float p1[8], p2[8];
#pragma unroll
    for (int v = 0; v < 8; ++v) {
        p1[v] = 0.f; p2[v] = 0.f;
#pragma unroll
        for (int ot = 0; ot < 4; ++ot) {
            p1[v] += acc[ot][v] * a1v[ot];
            p2[v] += acc[ot][v] * a2v[ot];
        }
    }
#pragma unroll
    for (int m = 1; m < 16; m <<= 1) {
#pragma unroll
        for (int v = 0; v < 8; ++v) {
            p1[v] += __shfl_xor(p1[v], m, 32);
            p2[v] += __shfl_xor(p2[v], m, 32);
        }
    }
    if (mrow == 0) {   // lanes 0 (rows 0-7) and 16 (rows 8-15)
#pragma unroll
        for (int v = 0; v < 8; ++v) {
            f1[r0 + v + 8 * hi] = p1[v];
            f2[r0 + v + 8 * hi] = p2[v];
        }
    }
}

// =====================================================================
// Kernel 2: fused masked-softmax attention, flash-style online softmax.
// One wave owns 16 query rows, streams 2048 cols in 64-wide tiles.
// P (bf16) @ Wh (bf16) -> f32 acc via v_wmma_f32_16x16x32_bf16.
// adj (128 MB) is read exactly once -> bandwidth-bound ~6 us floor.
// Accumulator rescale is ballot-skipped once the running max stabilizes.
// =====================================================================
__global__ void __launch_bounds__(128)
gat_attn_kernel(const int* __restrict__ adj,
                const __bf16* __restrict__ WhT,
                const float* __restrict__ f1, const float* __restrict__ f2,
                float* __restrict__ out)
{
    const int lane = threadIdx.x & 31;
    const int wave = threadIdx.x >> 5;
    const int tile = blockIdx.x * (blockDim.x >> 5) + wave;   // 0..1023
    const int r0   = tile * 16;
    const int b    = r0 >> 11;
    const int n0   = r0 & (NNODE - 1);
    const int mrow = lane & 15;
    const int hi   = lane >> 4;

    const float f1m = f1[r0 + mrow];          // this lane's query-row bias
    const float* f2b = f2 + (size_t)b * NNODE;
    const int* adj_row = adj + ((size_t)b * NNODE + n0 + mrow) * NNODE;

    float mrun = -__builtin_inff();
    float lrun = 0.0f;
    v8f acc[4];
#pragma unroll
    for (int ot = 0; ot < 4; ++ot) acc[ot] = zero8();

    for (int c0 = 0; c0 < NNODE; c0 += 64) {
        // stream next adj tile into cache while we compute
        __builtin_prefetch(adj_row + c0 + 64 + hi * 32, 0, 1);

        // ---- e = mask(leaky(f1+f2)) for this lane's 32 columns ----
        // layout matches bf16 A-fragment: kf selects 32-col fragment,
        // g selects the K/K+16 half, hi selects lo/hi lane K offset.
        float ev[2][16];
        float tmax = -__builtin_inff();
#pragma unroll
        for (int kf = 0; kf < 2; ++kf) {
#pragma unroll
            for (int g = 0; g < 2; ++g) {
                const int cc = c0 + kf * 32 + g * 16 + hi * 8;
                const int4  m0 = *(const int4*)(adj_row + cc);
                const int4  m1 = *(const int4*)(adj_row + cc + 4);
                const float4 s0 = *(const float4*)(f2b + cc);
                const float4 s1 = *(const float4*)(f2b + cc + 4);
                const int   mi[8] = { m0.x, m0.y, m0.z, m0.w, m1.x, m1.y, m1.z, m1.w };
                const float sj[8] = { s0.x, s0.y, s0.z, s0.w, s1.x, s1.y, s1.z, s1.w };
#pragma unroll
                for (int j = 0; j < 8; ++j) {
                    const float t = f1m + sj[j];
                    float e = fmaxf(t, ALPHA * t);            // LeakyReLU
                    e = (mi[j] > 0) ? e : NEG_BIG;            // adjacency mask
                    ev[kf][g * 8 + j] = e;
                    tmax = fmaxf(tmax, e);
                }
            }
        }

        // ---- online softmax bookkeeping (row = lane pair l, l^16) ----
        tmax = fmaxf(tmax, __shfl_xor(tmax, 16, 32));
        const float mnew = fmaxf(mrun, tmax);

        // wave-uniform skip: rescale only if some row's max moved
        if (__ballot(mnew > mrun)) {
            const float corr = EXP2F((mrun - mnew) * LOG2E);  // 0 on first tile
            float sc[8];                                      // row M=v+8*hi
#pragma unroll
            for (int v = 0; v < 8; ++v) sc[v] = __shfl(corr, v + 8 * hi, 32);
#pragma unroll
            for (int ot = 0; ot < 4; ++ot)
#pragma unroll
                for (int v = 0; v < 8; ++v) acc[ot][v] *= sc[v];
            lrun *= corr;
            mrun = mnew;
        }

        float psum = 0.0f;
#pragma unroll
        for (int kf = 0; kf < 2; ++kf) {
            // ---- P fragment (bf16 A-layout) ----
            v16f pf;
#pragma unroll
            for (int j = 0; j < 16; ++j) {
                const float p = EXP2F((ev[kf][j] - mrun) * LOG2E);
                pf[j] = p;
                psum += p;
            }
            const v16bf P = __builtin_convertvector(pf, v16bf);

            // ---- B fragments from WhT: K contiguous in memory ----
#pragma unroll
            for (int ot = 0; ot < 4; ++ot) {
                const int o = mrow + ot * 16;
                const __bf16* wp =
                    WhT + ((size_t)(b * FOUT + o)) * NNODE + c0 + kf * 32 + hi * 8;
                BF16x16 Bf;
                Bf.q[0] = *(const uint4*)wp;
                Bf.q[1] = *(const uint4*)(wp + 16);
                acc[ot] = __builtin_amdgcn_wmma_f32_16x16x32_bf16(
                    false, P, false, Bf.v, (short)0, acc[ot], false, false);
            }
        }
        lrun += psum;
    }

    // ---- epilogue: normalize, ELU, store ----
    const float ltot = lrun + __shfl_xor(lrun, 16, 32);
    const float rinv = 1.0f / ltot;
    float linv[8];
#pragma unroll
    for (int v = 0; v < 8; ++v) linv[v] = __shfl(rinv, v + 8 * hi, 32);

#pragma unroll
    for (int ot = 0; ot < 4; ++ot) {
        const int o = mrow + ot * 16;
#pragma unroll
        for (int v = 0; v < 8; ++v) {
            float val = acc[ot][v] * linv[v];
            val = (val > 0.f) ? val : (EXP2F(val * LOG2E) - 1.0f);   // ELU
            out[((size_t)(b * NNODE + n0 + v + 8 * hi)) * FOUT + o] = val;
        }
    }
}

// =====================================================================
extern "C" void kernel_launch(void* const* d_in, const int* in_sizes, int n_in,
                              void* d_out, int out_size, void* d_ws, size_t ws_size,
                              hipStream_t stream) {
    const float* x   = (const float*)d_in[0];   // (8,2048,128) f32
    const int*   adj = (const int*)d_in[1];     // (8,2048,2048) i32
    const float* W   = (const float*)d_in[2];   // (64,128) f32
    const float* a   = (const float*)d_in[3];   // (128,1) f32
    float*       out = (float*)d_out;           // (8,2048,64) f32

    // workspace: WhT bf16 (2 MB) + f1 (64 KB) + f2 (64 KB) + Wbf (16 KB)
    __bf16* WhT = (__bf16*)d_ws;
    float* f1 = (float*)((char*)d_ws + (size_t)BATCH * FOUT * NNODE * 2);
    float* f2 = f1 + (size_t)BATCH * NNODE;
    __bf16* Wbf = (__bf16*)(f2 + (size_t)BATCH * NNODE);

    gat_cvtw_kernel<<<16, 256, 0, stream>>>(W, Wbf);
    gat_wh_kernel<<<128, 256, 0, stream>>>(x, Wbf, a, WhT, f1, f2);
    gat_attn_kernel<<<256, 128, 0, stream>>>(adj, WhT, f1, f2, out);
}
